// Rwkv7SelfAttention_26740466385135
// MI455X (gfx1250) — matrix operand order repaired and checked by reading, not compiled, not run on hardware
//
#include <hip/hip_runtime.h>
#include <hip/hip_bf16.h>
#include <math.h>

// RWKV-7 block for MI455X (gfx1250, wave32, WMMA).
// Big GEMMs: bf16 WMMA 16x16x32, f32 accumulate, 4x2 wmma tiles per wave
// (64x32), float4 global loads, token-shift mix fused at compile time,
// branch-free speculative prefetch. Scan: software-pipelined loads.

typedef __attribute__((ext_vector_type(16))) __bf16 v16bf;
typedef __attribute__((ext_vector_type(8)))  float  v8f;
typedef __attribute__((ext_vector_type(4)))  float  f4;

static constexpr int T  = 2048;
static constexpr int D  = 2048;
static constexpr int H  = 32;
static constexpr int NH = 64;        // head dim
static constexpr int TD = T * D;

__device__ __forceinline__ float sigmoidf_(float x) { return 1.0f / (1.0f + __expf(-x)); }

// ---------------------------------------------------------------- LayerNorm
__global__ __launch_bounds__(256) void k_ln(const float* __restrict__ x,
                                            const float* __restrict__ w,
                                            const float* __restrict__ b,
                                            float* __restrict__ xn) {
  const int t = blockIdx.x, tid = threadIdx.x;
  const float* xr = x + (size_t)t * D;
  __shared__ float red[256];
  float s = 0.f;
#pragma unroll
  for (int r = 0; r < D / 256; ++r) s += xr[tid + r * 256];
  red[tid] = s; __syncthreads();
  for (int o = 128; o > 0; o >>= 1) { if (tid < o) red[tid] += red[tid + o]; __syncthreads(); }
  const float mean = red[0] * (1.0f / D);
  __syncthreads();
  float vs = 0.f;
#pragma unroll
  for (int r = 0; r < D / 256; ++r) { float d0 = xr[tid + r * 256] - mean; vs += d0 * d0; }
  red[tid] = vs; __syncthreads();
  for (int o = 128; o > 0; o >>= 1) { if (tid < o) red[tid] += red[tid + o]; __syncthreads(); }
  const float rstd = rsqrtf(red[0] * (1.0f / D) + 1e-5f);
  float* on = xn + (size_t)t * D;
#pragma unroll
  for (int r = 0; r < D / 256; ++r) {
    const int c = tid + r * 256;
    on[c] = (xr[c] - mean) * rstd * w[c] + b[c];
  }
}

// ------------------------------------------------- token shift: sx = prev - xn
__global__ __launch_bounds__(256) void k_shift(const float* __restrict__ xn,
                                               const float* __restrict__ state1,
                                               float* __restrict__ sx,
                                               float* __restrict__ xn_last) {
  const int idx = blockIdx.x * 256 + threadIdx.x;
  const int t = idx >> 11;          // /D
  const int d = idx & (D - 1);
  const float cur = xn[idx];
  const float prev = (t == 0) ? state1[d] : xn[idx - D];
  sx[idx] = prev - cur;
  if (t == T - 1) xn_last[d] = cur;
}

// ---------------------------------------------------------- WMMA GEMM (NT)
// C[m,n] = sum_k (A[m,k] [+ As[m,k]*coef[k]]) * B[n,k]   [+ resid[m,n]]
// Block: 256 thr = 8 waves (2x4), wave tile 64x32 = 4x2 wmma 16x16x32 bf16.
// Block tile 128x128.
template <bool MIX, bool RES>
__global__ __launch_bounds__(256) void k_gemm(
    const float* __restrict__ A, const float* __restrict__ As,
    const float* __restrict__ coef, const float* __restrict__ B,
    const float* __restrict__ resid, float* __restrict__ C,
    int M, int Nc, int K) {
  const int tid  = threadIdx.x;
  const int wave = tid >> 5;
  const int lane = tid & 31;
  const int tileM0 = blockIdx.y * 128 + (wave >> 2) * 64;
  const int tileN0 = blockIdx.x * 128 + (wave & 3)  * 32;
  const int am   = lane & 15;      // A row / B col within 16
  const int agrp = lane >> 4;      // lane group (K interleave)

  v8f acc[4][2] = {};

  for (int k0 = 0; k0 < K; k0 += 32) {
    // A fragment layout (16x32 bf16): lane m=am; elements 0-7 hold
    // K = agrp*8 + 0..7, elements 8-15 hold K = 16 + agrp*8 + 0..7.
    // Both runs are 8 contiguous floats -> 2x float4 each.
    f4 c0, c1, c2, c3;
    if (MIX) {
      const f4* Cp = (const f4*)(coef + k0);
      c0 = Cp[agrp * 2]; c1 = Cp[agrp * 2 + 1];
      c2 = Cp[4 + agrp * 2]; c3 = Cp[4 + agrp * 2 + 1];
    }
    v16bf af[4];
#pragma unroll
    for (int rr = 0; rr < 4; ++rr) {
      const size_t ro = (size_t)(tileM0 + rr * 16 + am) * K + k0;
      const f4* Ap = (const f4*)(A + ro);
      f4 a0 = Ap[agrp * 2], a1 = Ap[agrp * 2 + 1];
      f4 a2 = Ap[4 + agrp * 2], a3 = Ap[4 + agrp * 2 + 1];
      if (MIX) {
        const f4* Sp = (const f4*)(As + ro);
        f4 s0 = Sp[agrp * 2], s1 = Sp[agrp * 2 + 1];
        f4 s2 = Sp[4 + agrp * 2], s3 = Sp[4 + agrp * 2 + 1];
        a0 += s0 * c0; a1 += s1 * c1; a2 += s2 * c2; a3 += s3 * c3;
      }
#pragma unroll
      for (int e = 0; e < 4; ++e) {
        af[rr][e]      = (__bf16)a0[e];
        af[rr][4 + e]  = (__bf16)a1[e];
        af[rr][8 + e]  = (__bf16)a2[e];
        af[rr][12 + e] = (__bf16)a3[e];
      }
    }
    // B fragment (32x16 bf16): lane n=am; same K layout as A: elements 0-7
    // K = agrp*8 + 0..7, elements 8-15 K = 16 + agrp*8 + 0..7 -> 4x float4.
    v16bf bfg[2];
#pragma unroll
    for (int cc = 0; cc < 2; ++cc) {
      const f4* Bp = (const f4*)(B + (size_t)(tileN0 + cc * 16 + am) * K + k0 + agrp * 8);
      f4 b0 = Bp[0], b1 = Bp[1], b2 = Bp[4], b3 = Bp[5];
#pragma unroll
      for (int e = 0; e < 4; ++e) {
        bfg[cc][e]      = (__bf16)b0[e];
        bfg[cc][4 + e]  = (__bf16)b1[e];
        bfg[cc][8 + e]  = (__bf16)b2[e];
        bfg[cc][12 + e] = (__bf16)b3[e];
      }
    }
    // Branch-free speculative prefetch two K tiles ahead (k0+64) (global_prefetch_b8;
    // translation failures are silently dropped, no LOADcnt cost).
    __builtin_prefetch(A + (size_t)(tileM0 + am) * K + k0 + 64, 0, 1);
    __builtin_prefetch(B + (size_t)(tileN0 + am) * K + k0 + 64, 0, 1);
#pragma unroll
    for (int mi = 0; mi < 4; ++mi)
#pragma unroll
      for (int ni = 0; ni < 2; ++ni)
        acc[mi][ni] = __builtin_amdgcn_wmma_f32_16x16x32_bf16(
            false, af[mi], false, bfg[ni], (short)0, acc[mi][ni], false, false);
  }

  // C/D layout: lane&15 = N, VGPR r -> M = (lane>>4)*8 + r
  const int cn = lane & 15;
  const int cm = (lane >> 4) * 8;
#pragma unroll
  for (int mi = 0; mi < 4; ++mi)
#pragma unroll
    for (int ni = 0; ni < 2; ++ni)
#pragma unroll
      for (int r = 0; r < 8; ++r) {
        const size_t o = (size_t)(tileM0 + mi * 16 + cm + r) * Nc + tileN0 + ni * 16 + cn;
        float v = acc[mi][ni][r];
        if (RES) v += resid[o];
        C[o] = v;
      }
}

// --------------------------------------------------------- LoRA down-proj
// H[t,f] = act( sum_d (xn+sx*coef)[t,d] * B1[d,f] ),  act: 0 none, 1 tanh, 2 sigmoid
__global__ __launch_bounds__(128) void k_lora_down(
    const float* __restrict__ xn, const float* __restrict__ sx,
    const float* __restrict__ coef, const float* __restrict__ B1,
    float* __restrict__ Hout, int F, int act) {
  const int t = blockIdx.x, tid = threadIdx.x;
  __shared__ float sa[256];
  const float* xr = xn + (size_t)t * D;
  const float* sr = sx + (size_t)t * D;
  float acc = 0.f;
  for (int d0 = 0; d0 < D; d0 += 256) {
#pragma unroll
    for (int u = 0; u < 2; ++u) {
      const int j = tid + u * 128;
      sa[j] = xr[d0 + j] + sr[d0 + j] * coef[d0 + j];
    }
    __syncthreads();
    if (tid < F) {
#pragma unroll 8
      for (int dd = 0; dd < 256; ++dd) acc += sa[dd] * B1[(size_t)(d0 + dd) * F + tid];
    }
    __syncthreads();
  }
  if (tid < F) {
    if (act == 1) acc = tanhf(acc);
    else if (act == 2) acc = sigmoidf_(acc);
    Hout[(size_t)t * F + tid] = acc;
  }
}

// ----------------------------------------------------------- LoRA up-proj
// mode 0: g path  Out = acc
// mode 1: w path  Out = exp(-0.606531*sigmoid(acc+bias))
// mode 2: a path  Out = sigmoid(acc+bias)
// mode 3: v path  Out = v + (v_first - v)*sigmoid(acc+bias)   (Out holds v_raw)
__global__ __launch_bounds__(256) void k_lora_up(
    const float* __restrict__ Hin, const float* __restrict__ B2,
    const float* __restrict__ bias, const float* __restrict__ vfirst,
    float* __restrict__ Out, int F, int mode) {
  const int t = blockIdx.x, tid = threadIdx.x;
  __shared__ float sh[128];
  if (tid < F) sh[tid] = Hin[(size_t)t * F + tid];
  __syncthreads();
#pragma unroll
  for (int u = 0; u < D / 256; ++u) {
    const int d = tid + u * 256;
    float acc = 0.f;
    for (int f = 0; f < F; ++f) acc += sh[f] * B2[(size_t)f * D + d];
    float o;
    if (mode == 0)      o = acc;
    else if (mode == 1) o = __expf(-0.606531f * sigmoidf_(acc + bias[d]));
    else if (mode == 2) o = sigmoidf_(acc + bias[d]);
    else {
      const float vr = Out[(size_t)t * D + d];
      o = vr + (vfirst[(size_t)t * D + d] - vr) * sigmoidf_(acc + bias[d]);
    }
    Out[(size_t)t * D + d] = o;
  }
}

// ---------------------------------- kk = normalize(k*k_k), k' = k*(1+(a-1)*k_a)
__global__ __launch_bounds__(64) void k_keys(
    float* __restrict__ k, float* __restrict__ kkout,
    const float* __restrict__ a, const float* __restrict__ k_k,
    const float* __restrict__ k_a) {
  const int b = blockIdx.x, i = threadIdx.x;       // b = t*H + h
  const size_t base = (size_t)b * NH;
  const int col = (b & (H - 1)) * NH + i;
  __shared__ float red[64];
  const float kr = k[base + i];
  const float kh = kr * k_k[col];
  red[i] = kh * kh; __syncthreads();
  for (int o = 32; o > 0; o >>= 1) { if (i < o) red[i] += red[i + o]; __syncthreads(); }
  const float nrm = sqrtf(red[0]) + 1e-6f;
  kkout[base + i] = kh / nrm;
  k[base + i] = kr * (1.0f + (a[base + i] - 1.0f) * k_a[col]);
}

// ------------------------------------------------------ sequential WKV scan
// 1 block per head, 64 lanes; lane i owns state row i (64 f32 in VGPRs).
// Software-pipelined: step t+1 global loads issue right after the publish
// barrier of step t, so their latency overlaps the 128-FMA update chain.
__global__ __launch_bounds__(64) void k_scan(
    const float* __restrict__ r, const float* __restrict__ w,
    const float* __restrict__ k, const float* __restrict__ v,
    const float* __restrict__ kk, const float* __restrict__ a,
    const float* __restrict__ s2in, float* __restrict__ xo,
    float* __restrict__ s2out) {
  const int h = blockIdx.x, i = threadIdx.x;
  float S[NH];
  const size_t sbase = (size_t)h * NH * NH + (size_t)i * NH;
#pragma unroll
  for (int j = 0; j < NH; ++j) S[j] = s2in[sbase + j];
  __shared__ float sr[NH], sw[NH], sk[NH], sv[NH], skk[NH], sab[NH];
  const size_t lane0 = (size_t)h * NH + i;
  // preload t = 0
  float lr = r[lane0], lw = w[lane0], lk = k[lane0];
  float lv = v[lane0], lkk = kk[lane0], la = a[lane0];
  for (int t = 0; t < T; ++t) {
    sr[i] = lr; sw[i] = lw; sk[i] = lk; sv[i] = lv;
    skk[i] = lkk; sab[i] = lkk * la;
    __syncthreads();
    // issue next step's loads now; waited on only at next iteration's publish
    if (t + 1 < T) {
      const size_t nb = lane0 + (size_t)(t + 1) * D;
      lr = r[nb]; lw = w[nb]; lk = k[nb];
      lv = v[nb]; lkk = kk[nb]; la = a[nb];
    }
    float sa = 0.f;
#pragma unroll
    for (int j = 0; j < NH; ++j) sa -= S[j] * skk[j];   // S @ (-kk)
    const float vi = sv[i];
    float out = 0.f;
#pragma unroll
    for (int j = 0; j < NH; ++j) {
      const float sj = S[j] * sw[j] + sa * sab[j] + vi * sk[j];
      S[j] = sj;
      out += sj * sr[j];                                // S @ r
    }
    xo[lane0 + (size_t)t * D] = out;
    __syncthreads();
  }
#pragma unroll
  for (int j = 0; j < NH; ++j) s2out[sbase + j] = S[j];
}

// ---------------------- per-head LN(eps=0.00064) + affine + r_k bonus + gate g
__global__ __launch_bounds__(64) void k_post(
    float* __restrict__ xo, const float* __restrict__ r,
    const float* __restrict__ k, const float* __restrict__ v,
    const float* __restrict__ g, const float* __restrict__ r_k,
    const float* __restrict__ lnw, const float* __restrict__ lnb) {
  const int b = blockIdx.x, i = threadIdx.x;       // b = t*H + h
  const size_t base = (size_t)b * NH;
  const int col = (b & (H - 1)) * NH + i;
  __shared__ float red[64];
  const float xv = xo[base + i];
  red[i] = xv; __syncthreads();
  for (int o = 32; o > 0; o >>= 1) { if (i < o) red[i] += red[i + o]; __syncthreads(); }
  const float mean = red[0] * (1.0f / NH);
  __syncthreads();
  const float dv = xv - mean;
  red[i] = dv * dv; __syncthreads();
  for (int o = 32; o > 0; o >>= 1) { if (i < o) red[i] += red[i + o]; __syncthreads(); }
  const float rstd = rsqrtf(red[0] * (1.0f / NH) + 0.00064f);
  __syncthreads();
  red[i] = r[base + i] * k[base + i] * r_k[col]; __syncthreads();
  for (int o = 32; o > 0; o >>= 1) { if (i < o) red[i] += red[i + o]; __syncthreads(); }
  const float bonus = red[0] * v[base + i];
  xo[base + i] = ((xv - mean) * rstd * lnw[col] + lnb[col] + bonus) * g[base + i];
}

// --------------------------------------------------------------- launcher
extern "C" void kernel_launch(void* const* d_in, const int* in_sizes, int n_in,
                              void* d_out, int out_size, void* d_ws, size_t ws_size,
                              hipStream_t stream) {
  (void)in_sizes; (void)n_in; (void)out_size; (void)ws_size;
  const float* x       = (const float*)d_in[0];
  const float* state1  = (const float*)d_in[1];
  const float* state2  = (const float*)d_in[2];
  const float* v_first = (const float*)d_in[3];
  const float* ln1_w   = (const float*)d_in[4];
  const float* ln1_b   = (const float*)d_in[5];
  const float* x_r = (const float*)d_in[6];
  const float* x_w = (const float*)d_in[7];
  const float* x_k = (const float*)d_in[8];
  const float* x_v = (const float*)d_in[9];
  const float* x_a = (const float*)d_in[10];
  const float* x_g = (const float*)d_in[11];
  const float* W_r = (const float*)d_in[12];
  const float* W_k = (const float*)d_in[13];
  const float* W_v = (const float*)d_in[14];
  const float* W_o = (const float*)d_in[15];
  const float* w1  = (const float*)d_in[16];
  const float* w2  = (const float*)d_in[17];
  const float* w0  = (const float*)d_in[18];
  const float* a1  = (const float*)d_in[19];
  const float* a2  = (const float*)d_in[20];
  const float* a0  = (const float*)d_in[21];
  const float* v1  = (const float*)d_in[22];
  const float* v2  = (const float*)d_in[23];
  const float* v0  = (const float*)d_in[24];
  const float* g1  = (const float*)d_in[25];
  const float* g2  = (const float*)d_in[26];
  const float* k_k = (const float*)d_in[27];
  const float* k_a = (const float*)d_in[28];
  const float* r_k = (const float*)d_in[29];
  const float* ln_x_w = (const float*)d_in[30];
  const float* ln_x_b = (const float*)d_in[31];

  float* out0    = (float*)d_out;          // [T,D]
  float* out_xn  = out0 + (size_t)TD;      // [D]
  float* out_s2  = out_xn + D;             // [H,N,N]

  float* ws  = (float*)d_ws;
  float* xn  = ws + 0 * (size_t)TD;
  float* sx  = ws + 1 * (size_t)TD;
  float* rb  = ws + 2 * (size_t)TD;
  float* kb  = ws + 3 * (size_t)TD;
  float* vb  = ws + 4 * (size_t)TD;
  float* kkb = ws + 5 * (size_t)TD;
  float* wb  = ws + 6 * (size_t)TD;
  float* ab  = ws + 7 * (size_t)TD;
  float* gb  = ws + 8 * (size_t)TD;
  float* xob = ws + 9 * (size_t)TD;
  float* hw  = ws + 10 * (size_t)TD;
  float* ha  = hw + (size_t)T * 64;
  float* hv  = ha + (size_t)T * 64;
  float* hg  = hv + (size_t)T * 32;

  k_ln<<<T, 256, 0, stream>>>(x, ln1_w, ln1_b, xn);
  k_shift<<<TD / 256, 256, 0, stream>>>(xn, state1, sx, out_xn);

  const dim3 gg(D / 128, T / 128);
  k_gemm<true, false><<<gg, 256, 0, stream>>>(xn, sx, x_r, W_r, nullptr, rb, T, D, D);
  k_gemm<true, false><<<gg, 256, 0, stream>>>(xn, sx, x_k, W_k, nullptr, kb, T, D, D);
  k_gemm<true, false><<<gg, 256, 0, stream>>>(xn, sx, x_v, W_v, nullptr, vb, T, D, D);

  k_lora_down<<<T, 128, 0, stream>>>(xn, sx, x_w, w1, hw, 64, 1);   // tanh
  k_lora_down<<<T, 128, 0, stream>>>(xn, sx, x_a, a1, ha, 64, 0);
  k_lora_down<<<T, 128, 0, stream>>>(xn, sx, x_v, v1, hv, 32, 0);
  k_lora_down<<<T, 128, 0, stream>>>(xn, sx, x_g, g1, hg, 128, 2);  // sigmoid

  k_lora_up<<<T, 256, 0, stream>>>(hw, w2, w0, nullptr, wb, 64, 1);      // decay w
  k_lora_up<<<T, 256, 0, stream>>>(ha, a2, a0, nullptr, ab, 64, 2);      // a
  k_lora_up<<<T, 256, 0, stream>>>(hv, v2, v0, v_first, vb, 32, 3);      // v blend
  k_lora_up<<<T, 256, 0, stream>>>(hg, g2, nullptr, nullptr, gb, 128, 0);// g

  k_keys<<<T * H, 64, 0, stream>>>(kb, kkb, ab, k_k, k_a);
  k_scan<<<H, 64, 0, stream>>>(rb, wb, kb, vb, kkb, ab, state2, xob, out_s2);
  k_post<<<T * H, 64, 0, stream>>>(xob, rb, kb, vb, gb, r_k, ln_x_w, ln_x_b);

  // out = x + xo @ W_o.T  (residual fused into WMMA epilogue)
  k_gemm<false, true><<<gg, 256, 0, stream>>>(xob, xob, nullptr, W_o, x, out0, T, D, D);
}
